// AttentionLayer_69707319214359
// MI455X (gfx1250) — matrix operand order repaired and checked by reading, not compiled, not run on hardware
//
#include <hip/hip_runtime.h>

// ---------------------------------------------------------------------------
// MI455X / gfx1250, wave32. All matmuls via v_wmma_f32_16x16x32_bf16.
// Shapes: B=2, S=2048, E=2048, H=16, D=128, 3E=6144.
// GEMMs: LDS-tiled + double-buffered, 8 waves/WG, 128x128 C block,
// 2(M) x 4(N) register blocking per wave (8 WMMA / k-step / wave).
// Kernel 3 stages its bf16 A tile with global_load_async_to_lds_b128.
// ---------------------------------------------------------------------------

typedef __attribute__((ext_vector_type(16))) __bf16 v16bf;
typedef __attribute__((ext_vector_type(8)))  __bf16 v8bf;
typedef __attribute__((ext_vector_type(8)))  float  v8f;
typedef __attribute__((ext_vector_type(4)))  float  v4f;

#define WMMA_BF16(a, b, c) \
  __builtin_amdgcn_wmma_f32_16x16x32_bf16(false, (a), false, (b), (short)0, (c), false, false)

// Native f32 -> bf16 conversion (hardware v_cvt on gfx1250).
static __device__ __forceinline__ __bf16 f2bf(float f) { return (__bf16)f; }
static __device__ __forceinline__ unsigned short f2bf_bits(float f) {
  return __builtin_bit_cast(unsigned short, (__bf16)f);
}

static __device__ __forceinline__ v16bf load16(const unsigned short* p0,
                                               const unsigned short* p1) {
  v8bf a = *reinterpret_cast<const v8bf*>(p0);
  v8bf b = *reinterpret_cast<const v8bf*>(p1);
  return __builtin_shufflevector(a, b, 0, 1, 2, 3, 4, 5, 6, 7,
                                 8, 9, 10, 11, 12, 13, 14, 15);
}

static __device__ __forceinline__ v8bf pack8(v4f f0, v4f f1) {
  v8bf pk;
#pragma unroll
  for (int i = 0; i < 4; ++i) { pk[i] = f2bf(f0[i]); pk[4 + i] = f2bf(f1[i]); }
  return pk;
}

// ---------------------------------------------------------------------------
// Kernel 1: QKV projection. C[4096 x 6144] = X[4096 x 2048] @ W_attn + b.
// 8 waves -> C block 128(M) x 128(N); wave = 2 M-tiles x 4 N-tiles.
// Results scattered bf16 into Q [B,H,S,D], K [B,H,S,D], Vt [B,H,D,S].
// ---------------------------------------------------------------------------
__global__ __launch_bounds__(256) void qkv_gemm_kernel(
    const float* __restrict__ X, const float* __restrict__ W,
    const float* __restrict__ bias, unsigned short* __restrict__ qbuf,
    unsigned short* __restrict__ kbuf, unsigned short* __restrict__ vtbuf) {
  __shared__ alignas(16) unsigned short lsA[2][128 * 32];   // A block row-major
  __shared__ alignas(16) unsigned short lsBT[2][128 * 32];  // B block transposed

  const int tid  = threadIdx.x;
  const int lane = tid & 31;
  const int w    = tid >> 5;
  const int wm   = w >> 1;          // 0..3 -> M sub-block of 32 rows
  const int wn   = w & 1;           // 0..1 -> N half of 64
  const int nl   = lane & 15;
  const int hi   = lane >> 4;
  const int koffA = hi * 8;
  const int koffB = hi * 16;

  const int nbase = blockIdx.x * 128;  // N = 6144
  const int mbase = blockIdx.y * 128;  // M = 4096

  const int sArow = tid >> 1;          // 0..127
  const int sAcol = (tid & 1) * 16;    // 0 or 16
  const int sBrow = tid >> 3;          // 0..31 (k)
  const int sBcol = (tid & 7) * 16;    // 0..112 (n)

  auto stage = [&](int k0, int buf) {
    // A: 128x32 fp32 -> bf16, 64B/thread coalesced, two packed 16B LDS stores
    {
      const float* src = X + (size_t)(mbase + sArow) * 2048 + k0 + sAcol;
      v4f f0 = *reinterpret_cast<const v4f*>(src);
      v4f f1 = *reinterpret_cast<const v4f*>(src + 4);
      v4f f2 = *reinterpret_cast<const v4f*>(src + 8);
      v4f f3 = *reinterpret_cast<const v4f*>(src + 12);
      unsigned short* dst = &lsA[buf][sArow * 32 + sAcol];
      *reinterpret_cast<v8bf*>(dst)     = pack8(f0, f1);
      *reinterpret_cast<v8bf*>(dst + 8) = pack8(f2, f3);
    }
    // B: 32x128 fp32 -> bf16, stored transposed BT[n][k]
    {
      const float* src = W + (size_t)(k0 + sBrow) * 6144 + nbase + sBcol;
      v4f f0 = *reinterpret_cast<const v4f*>(src);
      v4f f1 = *reinterpret_cast<const v4f*>(src + 4);
      v4f f2 = *reinterpret_cast<const v4f*>(src + 8);
      v4f f3 = *reinterpret_cast<const v4f*>(src + 12);
#pragma unroll
      for (int i = 0; i < 4; ++i) {
        lsBT[buf][(sBcol + i) * 32 + sBrow]      = f2bf_bits(f0[i]);
        lsBT[buf][(sBcol + 4 + i) * 32 + sBrow]  = f2bf_bits(f1[i]);
        lsBT[buf][(sBcol + 8 + i) * 32 + sBrow]  = f2bf_bits(f2[i]);
        lsBT[buf][(sBcol + 12 + i) * 32 + sBrow] = f2bf_bits(f3[i]);
      }
    }
  };

  v8f acc[2][4];
#pragma unroll
  for (int g = 0; g < 2; ++g)
#pragma unroll
    for (int t = 0; t < 4; ++t)
      acc[g][t] = v8f{0.f, 0.f, 0.f, 0.f, 0.f, 0.f, 0.f, 0.f};

  stage(0, 0);
  for (int kb = 0; kb < 64; ++kb) {
    __syncthreads();
    const int cur = kb & 1;
    if (kb + 1 < 64) stage((kb + 1) * 32, cur ^ 1);
    const unsigned short* pa0 = &lsA[cur][(wm * 32 + nl) * 32 + koffA];
    const unsigned short* pa1 = pa0 + 16 * 32;
    v16bf a0 = load16(pa0, pa0 + 16);
    v16bf a1 = load16(pa1, pa1 + 16);
#pragma unroll
    for (int t = 0; t < 4; ++t) {
      const unsigned short* pb = &lsBT[cur][(wn * 64 + t * 16 + nl) * 32 + koffA];
      v16bf b = load16(pb, pb + 16);
      acc[0][t] = WMMA_BF16(a0, b, acc[0][t]);
      acc[1][t] = WMMA_BF16(a1, b, acc[1][t]);
    }
  }

  // Epilogue: destination pointer + stride resolved once per tile.
#pragma unroll
  for (int g = 0; g < 2; ++g) {
    const int row0 = mbase + wm * 32 + g * 16 + 8 * hi;  // rows row0..row0+7
    const int bidx = row0 >> 11;
    const int s0   = row0 & 2047;
#pragma unroll
    for (int t = 0; t < 4; ++t) {
      const int ncol  = nbase + wn * 64 + t * 16 + nl;
      const float bv  = bias[ncol];
      const int which = ncol >> 11;  // 0=Q 1=K 2=V
      const int e     = ncol & 2047;
      const int h     = e >> 7;
      const int d     = e & 127;
      unsigned short* p;
      int stride;
      if (which == 2) {  // Vt [B,H,D,S]
        p = vtbuf + ((size_t)(bidx * 16 + h) * 128 + d) * 2048 + s0;
        stride = 1;
      } else {           // Q/K [B,H,S,D]
        unsigned short* base = (which == 0) ? qbuf : kbuf;
        p = base + ((size_t)(bidx * 16 + h) * 2048 + s0) * 128 + d;
        stride = 128;
      }
#pragma unroll
      for (int r = 0; r < 8; ++r)
        p[r * stride] = f2bf_bits(acc[g][t][r] + bv);
    }
  }
}

// ---------------------------------------------------------------------------
// Kernel 2: causal flash attention, one wave per 16-query tile per (b,h).
// S^T = K @ Q^T and O^T = V^T @ P^T: probabilities (C-layout) become the
// B operand with one shfl_xor(16) half-swap — no LDS transpose.
// ---------------------------------------------------------------------------
__global__ __launch_bounds__(128) void attn_kernel(
    const unsigned short* __restrict__ qB, const unsigned short* __restrict__ kB,
    const unsigned short* __restrict__ vtB, unsigned short* __restrict__ ctx) {
  const int lane = threadIdx.x & 31;
  const int wid  = threadIdx.x >> 5;
  const int qt   = blockIdx.x * 4 + wid;  // 0..127
  const int bh   = blockIdx.y;            // 0..31
  const int nl   = lane & 15;
  const int hi   = lane >> 4;
  const int koffA = hi * 8;
  const int koffB = hi * 16;

  const unsigned short* Qb = qB + (size_t)bh * 2048 * 128;
  const unsigned short* Kb = kB + (size_t)bh * 2048 * 128;
  const unsigned short* Vt = vtB + (size_t)bh * 128 * 2048;
  const int qrow = qt * 16 + nl;

  v16bf qf[4];
#pragma unroll
  for (int dc = 0; dc < 4; ++dc) {
    const unsigned short* p = Qb + (size_t)qrow * 128 + dc * 32 + koffB;
    qf[dc] = load16(p, p + 8);
  }

  v8f oc[8];
#pragma unroll
  for (int c = 0; c < 8; ++c) oc[c] = v8f{0.f, 0.f, 0.f, 0.f, 0.f, 0.f, 0.f, 0.f};

  float mrun = -1e30f, lrun = 0.f;
  const float sm = 0.08838834764831845f;  // 1/sqrt(128)
  const int jmax = qt >> 1;

  for (int j = 0; j <= jmax; ++j) {
    const int kv0 = j * 32;
    v8f s0 = {0.f, 0.f, 0.f, 0.f, 0.f, 0.f, 0.f, 0.f};
    v8f s1 = {0.f, 0.f, 0.f, 0.f, 0.f, 0.f, 0.f, 0.f};
#pragma unroll
    for (int dc = 0; dc < 4; ++dc) {
      const unsigned short* kp = Kb + (size_t)(kv0 + nl) * 128 + dc * 32 + koffB;
      v16bf a0 = load16(kp, kp + 8);
      v16bf a1 = load16(kp + 16 * 128, kp + 16 * 128 + 8);
      s0 = WMMA_BF16(a0, qf[dc], s0);
      s1 = WMMA_BF16(a1, qf[dc], s1);
    }
    float p0[8], p1[8];
    float tmax = -1e30f;
#pragma unroll
    for (int r = 0; r < 8; ++r) {
      const int kvg0 = kv0 + r + 8 * hi;
      const int kvg1 = kvg0 + 16;
      float v0 = s0[r] * sm;
      float v1 = s1[r] * sm;
      if (kvg0 > qrow) v0 = -1e30f;
      if (kvg1 > qrow) v1 = -1e30f;
      p0[r] = v0; p1[r] = v1;
      tmax = fmaxf(tmax, fmaxf(v0, v1));
    }
    tmax = fmaxf(tmax, __shfl_xor(tmax, 16, 32));
    const float mnew  = fmaxf(mrun, tmax);
    const float alpha = __expf(mrun - mnew);
    float tsum = 0.f;
#pragma unroll
    for (int r = 0; r < 8; ++r) {
      p0[r] = __expf(p0[r] - mnew);
      p1[r] = __expf(p1[r] - mnew);
      tsum += p0[r] + p1[r];
    }
    tsum += __shfl_xor(tsum, 16, 32);
    lrun = lrun * alpha + tsum;
    mrun = mnew;
#pragma unroll
    for (int c = 0; c < 8; ++c)
#pragma unroll
      for (int r = 0; r < 8; ++r) oc[c][r] *= alpha;

    v16bf pb;
#pragma unroll
    for (int r = 0; r < 8; ++r) {
      const float o0 = __shfl_xor(p0[r], 16, 32);
      const float o1 = __shfl_xor(p1[r], 16, 32);
      pb[r]     = f2bf(hi == 0 ? p0[r] : o1);
      pb[8 + r] = f2bf(hi == 0 ? o0 : p1[r]);
    }
#pragma unroll
    for (int c = 0; c < 8; ++c) {
      const unsigned short* vp = Vt + (size_t)(c * 16 + nl) * 2048 + kv0 + koffB;
      v16bf va = load16(vp, vp + 8);
      oc[c] = WMMA_BF16(va, pb, oc[c]);
    }
  }

  const float rl = 1.0f / lrun;
  const int b = bh >> 4, h = bh & 15;
  const int tok = b * 2048 + qt * 16 + nl;
  unsigned short* op = ctx + ((size_t)tok * 16 + h) * 128 + 8 * hi;
#pragma unroll
  for (int c = 0; c < 8; ++c)
#pragma unroll
    for (int r = 0; r < 8; ++r)
      op[c * 16 + r] = f2bf_bits(oc[c][r] * rl);
}

// ---------------------------------------------------------------------------
// Kernel 3: out = ctx[4096 x 2048](bf16) @ W_proj(f32->bf16) + b_proj, f32 out.
// A staged with CDNA5 async copies (global_load_async_to_lds_b128, ASYNCcnt).
// ---------------------------------------------------------------------------
__global__ __launch_bounds__(256) void out_gemm_kernel(
    const unsigned short* __restrict__ ctx, const float* __restrict__ W,
    const float* __restrict__ bias, float* __restrict__ out) {
  __shared__ alignas(16) unsigned short lsA[2][128 * 32];
  __shared__ alignas(16) unsigned short lsBT[2][128 * 32];

  const int tid  = threadIdx.x;
  const int lane = tid & 31;
  const int w    = tid >> 5;
  const int wm   = w >> 1;
  const int wn   = w & 1;
  const int nl   = lane & 15;
  const int hi   = lane >> 4;
  const int koffA = hi * 8;
  const int koffB = hi * 16;

  const int nbase = blockIdx.x * 128;  // N = 2048
  const int mbase = blockIdx.y * 128;  // M = 4096

  const int sArow = tid >> 1;          // 0..127
  const int sAcol = (tid & 1) * 16;    // 0 or 16
  const int sBrow = tid >> 3;
  const int sBcol = (tid & 7) * 16;

  auto stage = [&](int k0, int buf) {
    // A: bf16 -> two async 16B DMAs into LDS (ASYNCcnt path, no VGPR round trip)
    {
      const unsigned voff =
          (unsigned)(((size_t)(mbase + sArow) * 2048 + k0 + sAcol) * 2);
      const unsigned ldsoff =
          (unsigned)(uintptr_t)&lsA[buf][sArow * 32 + sAcol];
      asm volatile("global_load_async_to_lds_b128 %0, %1, %2"
                   :: "v"(ldsoff), "v"(voff), "s"(ctx) : "memory");
      asm volatile("global_load_async_to_lds_b128 %0, %1, %2 offset:16"
                   :: "v"(ldsoff), "v"(voff), "s"(ctx) : "memory");
    }
    // B: 32x128 fp32 -> bf16, stored transposed BT[n][k]
    {
      const float* src = W + (size_t)(k0 + sBrow) * 2048 + nbase + sBcol;
      v4f f0 = *reinterpret_cast<const v4f*>(src);
      v4f f1 = *reinterpret_cast<const v4f*>(src + 4);
      v4f f2 = *reinterpret_cast<const v4f*>(src + 8);
      v4f f3 = *reinterpret_cast<const v4f*>(src + 12);
#pragma unroll
      for (int i = 0; i < 4; ++i) {
        lsBT[buf][(sBcol + i) * 32 + sBrow]      = f2bf_bits(f0[i]);
        lsBT[buf][(sBcol + 4 + i) * 32 + sBrow]  = f2bf_bits(f1[i]);
        lsBT[buf][(sBcol + 8 + i) * 32 + sBrow]  = f2bf_bits(f2[i]);
        lsBT[buf][(sBcol + 12 + i) * 32 + sBrow] = f2bf_bits(f3[i]);
      }
    }
  };

  v8f acc[2][4];
#pragma unroll
  for (int g = 0; g < 2; ++g)
#pragma unroll
    for (int t = 0; t < 4; ++t)
      acc[g][t] = v8f{0.f, 0.f, 0.f, 0.f, 0.f, 0.f, 0.f, 0.f};

  stage(0, 0);
  for (int kb = 0; kb < 64; ++kb) {
    asm volatile("s_wait_asynccnt 0x0" ::: "memory");
    __syncthreads();
    const int cur = kb & 1;
    if (kb + 1 < 64) stage((kb + 1) * 32, cur ^ 1);
    const unsigned short* pa0 = &lsA[cur][(wm * 32 + nl) * 32 + koffA];
    const unsigned short* pa1 = pa0 + 16 * 32;
    v16bf a0 = load16(pa0, pa0 + 16);
    v16bf a1 = load16(pa1, pa1 + 16);
#pragma unroll
    for (int t = 0; t < 4; ++t) {
      const unsigned short* pb = &lsBT[cur][(wn * 64 + t * 16 + nl) * 32 + koffA];
      v16bf b = load16(pb, pb + 16);
      acc[0][t] = WMMA_BF16(a0, b, acc[0][t]);
      acc[1][t] = WMMA_BF16(a1, b, acc[1][t]);
    }
  }

#pragma unroll
  for (int g = 0; g < 2; ++g) {
    const int row0 = mbase + wm * 32 + g * 16 + 8 * hi;
#pragma unroll
    for (int t = 0; t < 4; ++t) {
      const int ncol = nbase + wn * 64 + t * 16 + nl;
      const float bv = bias[ncol];
      float* p = out + (size_t)row0 * 2048 + ncol;
#pragma unroll
      for (int r = 0; r < 8; ++r)
        p[(size_t)r * 2048] = acc[g][t][r] + bv;
    }
  }
}

// ---------------------------------------------------------------------------
extern "C" void kernel_launch(void* const* d_in, const int* in_sizes, int n_in,
                              void* d_out, int out_size, void* d_ws, size_t ws_size,
                              hipStream_t stream) {
  (void)in_sizes; (void)n_in; (void)out_size; (void)ws_size;
  const float* X      = (const float*)d_in[0];  // [B,S,E]
  const float* W_attn = (const float*)d_in[1];  // [E,3E]
  const float* b_attn = (const float*)d_in[2];  // [3E]
  const float* W_proj = (const float*)d_in[3];  // [E,E]
  const float* b_proj = (const float*)d_in[4];  // [E]
  float* out = (float*)d_out;

  const size_t PLANE = (size_t)2 * 16 * 2048 * 128;  // 8,388,608 bf16 elements
  unsigned short* qb  = (unsigned short*)d_ws;
  unsigned short* kb  = qb + PLANE;
  unsigned short* vtb = kb + PLANE;
  unsigned short* ctx = vtb + PLANE;

  // QKV projection: grid = (6144/128, 4096/128)
  qkv_gemm_kernel<<<dim3(48, 32), 256, 0, stream>>>(X, W_attn, b_attn, qb, kb, vtb);
  // Flash attention: 128 q-tiles / 4 waves = 32 blocks, 32 (b,h) pairs
  attn_kernel<<<dim3(32, 32), 128, 0, stream>>>(qb, kb, vtb, ctx);
  // Output projection: grid = (2048/128, 4096/128)
  out_gemm_kernel<<<dim3(16, 32), 256, 0, stream>>>(ctx, W_proj, b_proj, out);
}
